// AttentiveModel_46162308498228
// MI455X (gfx1250) — compile-verified, matrix operand order checked by reading, not run on hardware
//
#include <hip/hip_runtime.h>
#include <hip/hip_bf16.h>

// CDNA5 / gfx1250, wave32. GEMMs on v_wmma_f32_16x16x32_f16 (fp16 in, fp32
// accumulate); elementwise/softmax math in fp32.
// 64x64 block tile, 4 wave32 per block, each wave owns a 32x32 quadrant
// (2x2 WMMA subtiles, fragment reuse). Staging: global_load_b128 ->
// v_cvt_pk_f16_f32 -> ds_store_b128 on the interior fast path, plus
// global_prefetch_b8 of the next K-tile.

typedef __attribute__((ext_vector_type(16))) _Float16 v16h;
typedef __attribute__((ext_vector_type(8)))  _Float16 v8h;
typedef __attribute__((ext_vector_type(8)))  float    v8f;

#define BT   64    // block tile (M and N)
#define KT   32    // K tile
#define LDSK 40    // padded row stride (80B): rows stay 16B aligned, banks spread

// Stage a 64x32 tile from row-major src (row stride ld floats) into
// dst[row][col]. 128 threads, 2 passes: row r=(tid>>2)+32*it, cols (tid&3)*8..+7
__device__ __forceinline__ void stage_rowmajor64(_Float16 (*dst)[LDSK],
                                                 const float* __restrict__ src,
                                                 long ld, int rows_avail,
                                                 int cols_avail, int tid)
{
#pragma unroll
    for (int it = 0; it < 2; ++it) {
        const int r  = (tid >> 2) + 32 * it;
        const int c0 = (tid & 3) * 8;
        if (r < rows_avail && c0 + 8 <= cols_avail) {
            const float4* p = reinterpret_cast<const float4*>(src + (long)r * ld + c0);
            float4 f0 = p[0], f1 = p[1];
            v8h h;
            h[0] = (_Float16)f0.x; h[1] = (_Float16)f0.y;
            h[2] = (_Float16)f0.z; h[3] = (_Float16)f0.w;
            h[4] = (_Float16)f1.x; h[5] = (_Float16)f1.y;
            h[6] = (_Float16)f1.z; h[7] = (_Float16)f1.w;
            *reinterpret_cast<v8h*>(&dst[r][c0]) = h;   // 16B aligned (80r + 2c0)
        } else {
#pragma unroll
            for (int j = 0; j < 8; ++j) {
                float v = (r < rows_avail && c0 + j < cols_avail)
                              ? src[(long)r * ld + c0 + j] : 0.0f;
                dst[r][c0 + j] = (_Float16)v;
            }
        }
    }
}

// Stage dst[n][k] = src[k*ld + n] (src row-major [K x N] slice, 32 k-rows x
// 64 n-cols) with a register transpose; coalesced along n.
__device__ __forceinline__ void stage_colmajor64(_Float16 (*dst)[LDSK],
                                                 const float* __restrict__ src,
                                                 long ld, int krows_avail,
                                                 int ncols_avail, int tid)
{
#pragma unroll
    for (int it = 0; it < 2; ++it) {
        const int k  = tid >> 2;
        const int n0 = (tid & 3) * 8 + 32 * it;
        if (k < krows_avail && n0 + 8 <= ncols_avail) {
            const float4* p = reinterpret_cast<const float4*>(src + (long)k * ld + n0);
            float4 f0 = p[0], f1 = p[1];
            float f[8] = {f0.x, f0.y, f0.z, f0.w, f1.x, f1.y, f1.z, f1.w};
#pragma unroll
            for (int j = 0; j < 8; ++j) dst[n0 + j][k] = (_Float16)f[j];
        } else {
#pragma unroll
            for (int j = 0; j < 8; ++j) {
                float v = (k < krows_avail && n0 + j < ncols_avail)
                              ? src[(long)k * ld + n0 + j] : 0.0f;
                dst[n0 + j][k] = (_Float16)v;
            }
        }
    }
}

// Build one 16x32 fragment (16-bit A/B lane layout) from LDS row-block base.
__device__ __forceinline__ v16h frag_from_lds(const _Float16 (*s)[LDSK],
                                              int row_base, int mr, int sel)
{
    v16h f;
#pragma unroll
    for (int i = 0; i < 8; ++i) {
        f[i]     = s[row_base + mr][sel * 8 + i];
        f[8 + i] = s[row_base + mr][16 + sel * 8 + i];
    }
    return f;
}

// ---------------------------------------------------------------------------
// Generic batched GEMM:  C = act(A @ B + bias)
//   A: [M,K] row-major (batch stride sA)
//   TRANSB==0 -> B [K,N] row-major ; TRANSB==1 -> B [N,K] row-major (A@B^T)
//   ACT: 0=none 1=relu 2=sigmoid
// ---------------------------------------------------------------------------
template <int ACT, int TRANSB>
__global__ void gemm_f16_kernel(const float* __restrict__ A,
                                const float* __restrict__ Bm,
                                const float* __restrict__ bias,
                                float* __restrict__ C,
                                int M, int N, int K,
                                long sA, long sB, long sC)
{
    __shared__ _Float16 As[BT][LDSK];   // As[m][k]
    __shared__ _Float16 Bs[BT][LDSK];   // Bs[n][k]

    const int bz = blockIdx.z;
    A  += (long)bz * sA;
    Bm += (long)bz * sB;
    C  += (long)bz * sC;

    const int tid  = threadIdx.x;
    const int lane = tid & 31;
    const int wave = tid >> 5;
    const int wm   = (wave >> 1) * 32;     // wave quadrant inside 64x64 tile
    const int wn   = (wave & 1)  * 32;
    const int m0   = blockIdx.y * BT;
    const int n0   = blockIdx.x * BT;
    const int mr   = lane & 15;
    const int sel  = lane >> 4;            // half-select per ISA 16-bit layout

    const float* Abase = A + (long)m0 * K;
    const int rowsA = M - m0;
    const int colsB = N - n0;

    v8f acc[2][2] = {};

    for (int kk = 0; kk < K; kk += KT) {
        const int kav = K - kk;

        // prefetch next K-tile while this one computes
        if (kk + KT < K) {
            __builtin_prefetch(Abase + kk + KT + (long)(tid >> 2) * K, 0, 0);
            if (TRANSB)
                __builtin_prefetch(Bm + (long)(n0 + (tid >> 2)) * K + kk + KT, 0, 0);
            else
                __builtin_prefetch(Bm + (long)(kk + KT + (tid >> 2)) * N + n0, 0, 0);
        }

        stage_rowmajor64(As, Abase + kk, K, rowsA, kav, tid);
        if (TRANSB)
            stage_rowmajor64(Bs, Bm + (long)n0 * K + kk, K, colsB, kav, tid);
        else
            stage_colmajor64(Bs, Bm + (long)kk * N + n0, N, kav, colsB, tid);
        __syncthreads();

        v16h af[2], bf[2];
#pragma unroll
        for (int t = 0; t < 2; ++t) {
            af[t] = frag_from_lds(As, wm + 16 * t, mr, sel);
            bf[t] = frag_from_lds(Bs, wn + 16 * t, mr, sel);
        }
#pragma unroll
        for (int ti = 0; ti < 2; ++ti)
#pragma unroll
            for (int tj = 0; tj < 2; ++tj)
                acc[ti][tj] = __builtin_amdgcn_wmma_f32_16x16x32_f16(
                    false, af[ti], false, bf[tj], (short)0, acc[ti][tj],
                    false, false);
        __syncthreads();
    }

#pragma unroll
    for (int ti = 0; ti < 2; ++ti)
#pragma unroll
        for (int tj = 0; tj < 2; ++tj)
#pragma unroll
            for (int r = 0; r < 8; ++r) {
                int gm = m0 + wm + 16 * ti + r + 8 * sel;
                int gn = n0 + wn + 16 * tj + mr;
                if (gm < M && gn < N) {
                    float v = acc[ti][tj][r];
                    if (bias) v += bias[gn];
                    if (ACT == 1)      v = fmaxf(v, 0.0f);
                    else if (ACT == 2) v = 1.0f / (1.0f + __expf(-v));
                    C[(long)gm * N + gn] = v;
                }
            }
}

// ---------------------------------------------------------------------------
// Compare GEMM with the [e, att, e-att, e*att] concat fused into the A loader.
//   K = 4*D; B: [4D, N] row-major.  C = relu(cat(e,att) @ B + bias)
// ---------------------------------------------------------------------------
__global__ void gemm_cat_f16_kernel(const float* __restrict__ e,
                                    const float* __restrict__ att,
                                    const float* __restrict__ Bm,
                                    const float* __restrict__ bias,
                                    float* __restrict__ C,
                                    int M, int N, int D)
{
    __shared__ _Float16 As[BT][LDSK];
    __shared__ _Float16 Bs[BT][LDSK];

    const int K    = 4 * D;
    const int tid  = threadIdx.x;
    const int lane = tid & 31;
    const int wave = tid >> 5;
    const int wm   = (wave >> 1) * 32;
    const int wn   = (wave & 1)  * 32;
    const int m0   = blockIdx.y * BT;
    const int n0   = blockIdx.x * BT;
    const int mr   = lane & 15;
    const int sel  = lane >> 4;

    v8f acc[2][2] = {};

    for (int kk = 0; kk < K; kk += KT) {
        // ---- A tile: concat segments computed on the fly ----
#pragma unroll
        for (int it = 0; it < 2; ++it) {
            const int r  = (tid >> 2) + 32 * it;
            const int c0 = (tid & 3) * 8;
            const int gm = m0 + r;
            const float* erow = e   + (long)gm * D;
            const float* arow = att + (long)gm * D;
#pragma unroll
            for (int j = 0; j < 8; ++j) {
                int gk = kk + c0 + j;
                float v = 0.0f;
                if (gm < M && gk < K) {
                    int seg = gk / D;
                    int d   = gk - seg * D;
                    float ev = erow[d];
                    if (seg == 0) v = ev;
                    else {
                        float av = arow[d];
                        v = (seg == 1) ? av : (seg == 2) ? (ev - av) : (ev * av);
                    }
                }
                As[r][c0 + j] = (_Float16)v;
            }
        }
        // ---- B tile (natural [K,N] layout, register transpose) ----
        stage_colmajor64(Bs, Bm + (long)kk * N + n0, N, K - kk, N - n0, tid);
        __syncthreads();

        v16h af[2], bf[2];
#pragma unroll
        for (int t = 0; t < 2; ++t) {
            af[t] = frag_from_lds(As, wm + 16 * t, mr, sel);
            bf[t] = frag_from_lds(Bs, wn + 16 * t, mr, sel);
        }
#pragma unroll
        for (int ti = 0; ti < 2; ++ti)
#pragma unroll
            for (int tj = 0; tj < 2; ++tj)
                acc[ti][tj] = __builtin_amdgcn_wmma_f32_16x16x32_f16(
                    false, af[ti], false, bf[tj], (short)0, acc[ti][tj],
                    false, false);
        __syncthreads();
    }

#pragma unroll
    for (int ti = 0; ti < 2; ++ti)
#pragma unroll
        for (int tj = 0; tj < 2; ++tj)
#pragma unroll
            for (int r = 0; r < 8; ++r) {
                int gm = m0 + wm + 16 * ti + r + 8 * sel;
                int gn = n0 + wn + 16 * tj + mr;
                if (gm < M && gn < N) {
                    float v = acc[ti][tj][r] + bias[gn];
                    C[(long)gm * N + gn] = fmaxf(v, 0.0f);
                }
            }
}

// ---------------------------------------------------------------------------
// Elementwise / support kernels (fp32)
// ---------------------------------------------------------------------------
__global__ void embed_kernel(const int* __restrict__ idx,
                             const float* __restrict__ emb,
                             float* __restrict__ out, long n_tok, int D)
{
    long i = (long)blockIdx.x * blockDim.x + threadIdx.x;
    long total = n_tok * D;
    if (i < total) {
        long t = i / D;
        int  d = (int)(i - t * D);
        out[i] = emb[(long)idx[t] * D + d];
    }
}

// x = t*h + (1-t)*x   (in place on x)
__global__ void highway_kernel(const float* __restrict__ h,
                               const float* __restrict__ t,
                               float* __restrict__ x, long n)
{
    long i = (long)blockIdx.x * blockDim.x + threadIdx.x;
    if (i < n) {
        float tv = t[i];
        x[i] = tv * h[i] + (1.0f - tv) * x[i];
    }
}

// sim[b][i][j] += sum_p 1/(1+|q1[b][i][p]-q2[b][j][p]|)
__global__ void dist_att_kernel(const float* __restrict__ q1,
                                const float* __restrict__ q2,
                                float* __restrict__ sim,
                                int S1, int S2, int P)
{
    __shared__ float q1row[256];
    const int b = blockIdx.y, i = blockIdx.x, j = threadIdx.x;
    for (int p = threadIdx.x; p < P; p += blockDim.x)
        q1row[p] = q1[((long)b * S1 + i) * P + p];
    __syncthreads();
    const float* q2p = q2 + ((long)b * S2 + j) * P;
    float s = 0.0f;
    for (int p = 0; p < P; ++p)
        s += 1.0f / (1.0f + fabsf(q1row[p] - q2p[p]));
    sim[((long)b * S1 + i) * S2 + j] += s;
}

// softmax over last axis; one 128-thread block per row of length S2==blockDim
__global__ void softmax_rows_kernel(const float* __restrict__ sim,
                                    float* __restrict__ out, int S2)
{
    __shared__ float red[128];
    const int row = blockIdx.x, j = threadIdx.x;
    float v = sim[(long)row * S2 + j];
    red[j] = v; __syncthreads();
    for (int s = 64; s > 0; s >>= 1) { if (j < s) red[j] = fmaxf(red[j], red[j + s]); __syncthreads(); }
    float mx = red[0]; __syncthreads();
    float e = __expf(v - mx);
    red[j] = e; __syncthreads();
    for (int s = 64; s > 0; s >>= 1) { if (j < s) red[j] += red[j + s]; __syncthreads(); }
    out[(long)row * S2 + j] = e / red[0];
}

// out[b][j][i] = softmax_i( sim[b][i][j] )   (softmax of sim^T rows)
__global__ void softmax_cols_kernel(const float* __restrict__ sim,
                                    float* __restrict__ out, int S1, int S2)
{
    __shared__ float red[128];
    const int b = blockIdx.y, j = blockIdx.x, i = threadIdx.x;
    float v = sim[((long)b * S1 + i) * S2 + j];
    red[i] = v; __syncthreads();
    for (int s = 64; s > 0; s >>= 1) { if (i < s) red[i] = fmaxf(red[i], red[i + s]); __syncthreads(); }
    float mx = red[0]; __syncthreads();
    float e = __expf(v - mx);
    red[i] = e; __syncthreads();
    for (int s = 64; s > 0; s >>= 1) { if (i < s) red[i] += red[i + s]; __syncthreads(); }
    out[((long)b * S2 + j) * S1 + i] = e / red[0];
}

// pooled[b] = concat(max_s v1, max_s v2, sum_s v1, sum_s v2)  -> [B, 4P]
__global__ void pool_kernel(const float* __restrict__ v1,
                            const float* __restrict__ v2,
                            float* __restrict__ out, int S, int P)
{
    const int b = blockIdx.x;
    for (int p = threadIdx.x; p < P; p += blockDim.x) {
        float mx1 = -3.402823466e+38f, mx2 = mx1, s1 = 0.0f, s2 = 0.0f;
        for (int s = 0; s < S; ++s) {
            float a = v1[((long)b * S + s) * P + p];
            float c = v2[((long)b * S + s) * P + p];
            mx1 = fmaxf(mx1, a); s1 += a;
            mx2 = fmaxf(mx2, c); s2 += c;
        }
        out[(long)b * 4 * P + p]         = mx1;
        out[(long)b * 4 * P + P + p]     = mx2;
        out[(long)b * 4 * P + 2 * P + p] = s1;
        out[(long)b * 4 * P + 3 * P + p] = s2;
    }
}

// ---------------------------------------------------------------------------
// Host orchestration
// ---------------------------------------------------------------------------
extern "C" void kernel_launch(void* const* d_in, const int* in_sizes, int n_in,
                              void* d_out, int out_size, void* d_ws, size_t ws_size,
                              hipStream_t stream)
{
    (void)in_sizes; (void)n_in; (void)out_size; (void)ws_size;

    const int Bc = 32, S = 128, D = 300, P = 200;
    const long MT = (long)Bc * S;            // 4096 rows per side

    const int*   x1   = (const int*)d_in[0];
    const int*   x2   = (const int*)d_in[1];
    const float* emb  = (const float*)d_in[2];
    const float* hw1_Wh = (const float*)d_in[3],  *hw1_bh = (const float*)d_in[4];
    const float* hw1_Wt = (const float*)d_in[5],  *hw1_bt = (const float*)d_in[6];
    const float* hw2_Wh = (const float*)d_in[7],  *hw2_bh = (const float*)d_in[8];
    const float* hw2_Wt = (const float*)d_in[9],  *hw2_bt = (const float*)d_in[10];
    const float* mul_W1 = (const float*)d_in[11], *mul_b1 = (const float*)d_in[12];
    const float* mul_W2 = (const float*)d_in[13], *mul_b2 = (const float*)d_in[14];
    const float* dis_W1 = (const float*)d_in[15], *dis_b1 = (const float*)d_in[16];
    const float* dis_W2 = (const float*)d_in[17], *dis_b2 = (const float*)d_in[18];
    const float* cmp_W1 = (const float*)d_in[19], *cmp_b1 = (const float*)d_in[20];
    const float* cmp_W2 = (const float*)d_in[21], *cmp_b2 = (const float*)d_in[22];
    const float* ch1_Wh = (const float*)d_in[23], *ch1_bh = (const float*)d_in[24];
    const float* ch1_Wt = (const float*)d_in[25], *ch1_bt = (const float*)d_in[26];
    const float* ch2_Wh = (const float*)d_in[27], *ch2_bh = (const float*)d_in[28];
    const float* ch2_Wt = (const float*)d_in[29], *ch2_bt = (const float*)d_in[30];
    const float* agg_W1 = (const float*)d_in[31], *agg_b1 = (const float*)d_in[32];
    const float* agg_W2 = (const float*)d_in[33], *agg_b2 = (const float*)d_in[34];
    const float* out_W  = (const float*)d_in[35], *out_b  = (const float*)d_in[36];

    // ---- workspace arena (~56 MB of fp32; hot set lives in the 192 MB L2) ----
    float* W = (float*)d_ws;
    size_t off = 0;
    auto alloc = [&](size_t n) { float* p = W + off; off += n; return p; };
    float* e1   = alloc(MT * D);
    float* e2   = alloc(MT * D);
    float* th   = alloc(MT * D);            // temp (h / ff hidden)
    float* tt   = alloc(MT * D);            // temp (t gate)
    float* p1   = alloc(MT * P);
    float* p2   = alloc(MT * P);
    float* q1   = alloc(MT * P);
    float* q2   = alloc(MT * P);
    float* sim  = alloc((long)Bc * S * S);
    float* srow = alloc((long)Bc * S * S);
    float* scol = alloc((long)Bc * S * S);
    float* alph = alloc(MT * D);
    float* beta = alloc(MT * D);
    float* v1   = alloc(MT * P);
    float* v2   = alloc(MT * P);
    float* pool = alloc((long)Bc * 4 * P);
    float* y1   = alloc((long)Bc * P);
    float* y2   = alloc((long)Bc * P);

    auto gemm = [&](const float* A, const float* Bm, const float* bias, float* C,
                    int M, int N, int K, int batch,
                    long sA, long sB, long sC, int act, int transB) {
        dim3 g((N + BT - 1) / BT, (M + BT - 1) / BT, batch);
        if (transB)
            gemm_f16_kernel<0, 1><<<g, 128, 0, stream>>>(A, Bm, bias, C, M, N, K, sA, sB, sC);
        else if (act == 0)
            gemm_f16_kernel<0, 0><<<g, 128, 0, stream>>>(A, Bm, bias, C, M, N, K, sA, sB, sC);
        else if (act == 1)
            gemm_f16_kernel<1, 0><<<g, 128, 0, stream>>>(A, Bm, bias, C, M, N, K, sA, sB, sC);
        else
            gemm_f16_kernel<2, 0><<<g, 128, 0, stream>>>(A, Bm, bias, C, M, N, K, sA, sB, sC);
    };

    auto highway = [&](float* x, int M, int Dd,
                       const float* Wh, const float* bh,
                       const float* Wt, const float* bt) {
        gemm(x, Wh, bh, th, M, Dd, Dd, 1, 0, 0, 0, /*relu*/1, 0);
        gemm(x, Wt, bt, tt, M, Dd, Dd, 1, 0, 0, 0, /*sigmoid*/2, 0);
        long n = (long)M * Dd;
        highway_kernel<<<dim3((unsigned)((n + 255) / 256)), 256, 0, stream>>>(th, tt, x, n);
    };

    auto ff = [&](const float* x, int M, int K,
                  const float* W1, const float* b1,
                  const float* W2, const float* b2, float* out) {
        gemm(x,  W1, b1, th,  M, P, K, 1, 0, 0, 0, 1, 0);
        gemm(th, W2, b2, out, M, P, P, 1, 0, 0, 0, 1, 0);
    };

    // 1) embedding gather
    {
        long n = MT * D;
        embed_kernel<<<dim3((unsigned)((n + 255) / 256)), 256, 0, stream>>>(x1, emb, e1, MT, D);
        embed_kernel<<<dim3((unsigned)((n + 255) / 256)), 256, 0, stream>>>(x2, emb, e2, MT, D);
    }

    // 2) shared highway stack on both sides
    highway(e1, (int)MT, D, hw1_Wh, hw1_bh, hw1_Wt, hw1_bt);
    highway(e1, (int)MT, D, hw2_Wh, hw2_bh, hw2_Wt, hw2_bt);
    highway(e2, (int)MT, D, hw1_Wh, hw1_bh, hw1_Wt, hw1_bt);
    highway(e2, (int)MT, D, hw2_Wh, hw2_bh, hw2_Wt, hw2_bt);

    // 3) mul / dist projections
    ff(e1, (int)MT, D, mul_W1, mul_b1, mul_W2, mul_b2, p1);
    ff(e2, (int)MT, D, mul_W1, mul_b1, mul_W2, mul_b2, p2);
    ff(e1, (int)MT, D, dis_W1, dis_b1, dis_W2, dis_b2, q1);
    ff(e2, (int)MT, D, dis_W1, dis_b1, dis_W2, dis_b2, q2);

    // 4) att1 = p1 @ p2^T  (batched, B^T path)
    gemm(p1, p2, nullptr, sim, S, S, P, Bc,
         (long)S * P, (long)S * P, (long)S * S, 0, 1);

    // 5) att2: distance attention accumulated into sim
    dist_att_kernel<<<dim3(S, Bc), 128, 0, stream>>>(q1, q2, sim, S, S, P);

    // 6) softmaxes (rows -> beta weights; cols -> alpha weights)
    softmax_rows_kernel<<<dim3(Bc * S), 128, 0, stream>>>(sim, srow, S);
    softmax_cols_kernel<<<dim3(S, Bc), 128, 0, stream>>>(sim, scol, S, S);

    // 7) contexts: beta = srow @ e2, alpha = scol @ e1 (batched)
    gemm(srow, e2, nullptr, beta, S, D, S, Bc,
         (long)S * S, (long)S * D, (long)S * D, 0, 0);
    gemm(scol, e1, nullptr, alph, S, D, S, Bc,
         (long)S * S, (long)S * D, (long)S * D, 0, 0);

    // 8) compare stage: fused-concat GEMM (K = 4D) then second FF layer
    {
        dim3 g((P + BT - 1) / BT, (unsigned)((MT + BT - 1) / BT));
        gemm_cat_f16_kernel<<<g, 128, 0, stream>>>(e1, beta, cmp_W1, cmp_b1, th,
                                                   (int)MT, P, D);
        gemm(th, cmp_W2, cmp_b2, v1, (int)MT, P, P, 1, 0, 0, 0, 1, 0);
        gemm_cat_f16_kernel<<<g, 128, 0, stream>>>(e2, alph, cmp_W1, cmp_b1, th,
                                                   (int)MT, P, D);
        gemm(th, cmp_W2, cmp_b2, v2, (int)MT, P, P, 1, 0, 0, 0, 1, 0);
    }

    // 9) compare highway stack
    highway(v1, (int)MT, P, ch1_Wh, ch1_bh, ch1_Wt, ch1_bt);
    highway(v1, (int)MT, P, ch2_Wh, ch2_bh, ch2_Wt, ch2_bt);
    highway(v2, (int)MT, P, ch1_Wh, ch1_bh, ch1_Wt, ch1_bt);
    highway(v2, (int)MT, P, ch2_Wh, ch2_bh, ch2_Wt, ch2_bt);

    // 10) pooling -> [B, 4P]
    pool_kernel<<<dim3(Bc), 128, 0, stream>>>(v1, v2, pool, S, P);

    // 11) aggregate MLP + output head
    gemm(pool, agg_W1, agg_b1, y1, Bc, P, 4 * P, 1, 0, 0, 0, 1, 0);
    gemm(y1,   agg_W2, agg_b2, y2, Bc, P, P,     1, 0, 0, 0, 1, 0);
    gemm(y2,   out_W,  out_b, (float*)d_out, Bc, 3, P, 1, 0, 0, 0, 0, 0);
}
